// GraphLearn_12945031430641
// MI455X (gfx1250) — compile-verified, matrix operand order checked by reading, not run on hardware
//
#include <hip/hip_runtime.h>
#include <hip/hip_bf16.h>

#define NN 1024
#define DD 128

typedef __attribute__((ext_vector_type(16))) _Float16 v16h;
typedef __attribute__((ext_vector_type(8)))  float    v8f;

union V16H { unsigned int u[8]; v16h h; };

__device__ __forceinline__ unsigned int pack_h2(float a, float b) {
    union { _Float16 h[2]; unsigned int u; } p;
    p.h[0] = (_Float16)a;
    p.h[1] = (_Float16)b;
    return p.u;
}

// pack two f32 diffs to f16x2 and clear both sign bits (|.| commutes with RNE)
__device__ __forceinline__ unsigned int pack_abs_h2(float a, float b) {
    return pack_h2(a, b) & 0x7FFF7FFFu;
}

// One 16x16xK=32 WMMA step: A = |xi - xj| abs-diffs (2 d-values x 16 j'),
// B = one-hot-column w weights. Exact weighted-L1 partial sum into f32 acc.
__device__ __forceinline__ void tile_step(const float* __restrict__ row0,
                                          const float* __restrict__ row1,
                                          int jb, int jb1, int jb2,
                                          float xi0, float xi1,
                                          const v16h& b, v8f& acc)
{
    float4 p0 = *(const float4*)(row0 + jb + jb1);   // d0, j' run 1 (4 cols)
    float4 q0 = *(const float4*)(row0 + jb + jb2);   // d0, j' run 2
    float4 p1 = *(const float4*)(row1 + jb + jb1);   // d1, j' run 1
    float4 q1 = *(const float4*)(row1 + jb + jb2);   // d1, j' run 2

    V16H aa;
    aa.u[0] = pack_abs_h2(xi0 - p0.x, xi1 - p1.x);
    aa.u[1] = pack_abs_h2(xi0 - p0.y, xi1 - p1.y);
    aa.u[2] = pack_abs_h2(xi0 - p0.z, xi1 - p1.z);
    aa.u[3] = pack_abs_h2(xi0 - p0.w, xi1 - p1.w);
    aa.u[4] = pack_abs_h2(xi0 - q0.x, xi1 - q1.x);
    aa.u[5] = pack_abs_h2(xi0 - q0.y, xi1 - q1.y);
    aa.u[6] = pack_abs_h2(xi0 - q0.z, xi1 - q1.z);
    aa.u[7] = pack_abs_h2(xi0 - q0.w, xi1 - q1.w);

    acc = __builtin_amdgcn_wmma_f32_16x16x32_f16(false, aa.h, false, b,
                                                 (short)0, acc, false, false);
}

// Block: 128 threads = 4 waves. Block tile: 64 rows (i) x 32 cols (j).
// Wave w handles rows [bi0 + 16w, +16), both 16-col j-tiles.
__global__ __launch_bounds__(128)
void GraphLearn_scores_kernel(const float* __restrict__ x,
                              const float* __restrict__ w,
                              const float* __restrict__ bias,
                              float* __restrict__ S)
{
    // xjT[d][j] for 32 j cols, row pitch 36 floats (144B = 16B-aligned rows)
    __shared__ __align__(16) float lds_xjT[DD * 36];
    // xiT[d][i] for 64 i rows, row pitch 68 floats (272B = 16B-aligned rows)
    __shared__ __align__(16) float lds_xiT[DD * 68];
    // packed f16 (w[2k], w[2k+1])
    __shared__ unsigned int lds_wpk[DD / 2];

    const int tid = threadIdx.x;
    const int bj0 = blockIdx.x * 32;   // 32 col tiles -> gridDim.x = 32
    const int bi0 = blockIdx.y * 64;   // 64 row tiles -> gridDim.y = 16

    // Load x_j tile, transposed (coalesced global reads: consecutive d).
    for (int idx = tid; idx < 32 * DD; idx += 128) {
        int j = idx >> 7;          // 0..31
        int d = idx & (DD - 1);    // 0..127
        lds_xjT[d * 36 + j] = x[(bj0 + j) * DD + d];
    }
    // Load x_i tile, transposed.
    for (int idx = tid; idx < 64 * DD; idx += 128) {
        int i = idx >> 7;          // 0..63
        int d = idx & (DD - 1);
        lds_xiT[d * 68 + i] = x[(bi0 + i) * DD + d];
    }
    if (tid < DD / 2) lds_wpk[tid] = pack_h2(w[2 * tid], w[2 * tid + 1]);
    __syncthreads();

    const int wv   = tid >> 5;      // wave id 0..3
    const int lane = tid & 31;
    const int c    = lane & 15;     // A row / B+C column handled by this lane
    const int hi   = lane >> 4;

    // B one-hot layout constants (which VGPR of which lane holds column c's
    // K-pair (2c, 2c+1)) — derived from the 16-bit B VGPR striping.
    const bool bact  = (hi == ((c >> 2) & 1));
    const int  bslot = (c & 3) + ((c >= 8) ? 4 : 0);

    // A j' runs for this lane half: hi=0 -> {0..3, 8..11}; hi=1 -> {4..7, 12..15}
    const int jb1 = 4 * hi;
    const int jb2 = 8 + 4 * hi;

    v8f acc0 = {};
    v8f acc1 = {};

    const int irow = wv * 16 + c;   // this lane's fixed i row within block

    for (int dp = 0; dp < DD / 2; ++dp) {
        const int d0 = 2 * dp;
        const float* xr0 = &lds_xiT[d0 * 68];
        const float* xr1 = &lds_xiT[(d0 + 1) * 68];
        const float xi0 = xr0[irow];            // conflict-free (16 consec banks)
        const float xi1 = xr1[irow];

        // Build one-hot B: two f16 halves (w[d0], w[d1]) in exactly one VGPR
        // of the two lanes owning each output column; zeros elsewhere.
        const unsigned int wpk = lds_wpk[dp];
        V16H bb;
#pragma unroll
        for (int v = 0; v < 8; ++v)
            bb.u[v] = (bact && (bslot == v)) ? wpk : 0u;

        const float* row0 = &lds_xjT[d0 * 36];
        const float* row1 = &lds_xjT[(d0 + 1) * 36];

        tile_step(row0, row1,  0, jb1, jb2, xi0, xi1, bb.h, acc0);
        tile_step(row0, row1, 16, jb1, jb2, xi0, xi1, bb.h, acc1);
    }

    // Fused bias + relu, write raw (pre-softmax) scores.
    const float b0 = bias[0];
#pragma unroll
    for (int r = 0; r < 8; ++r) {
        const int row = bi0 + wv * 16 + r + 8 * hi;   // C layout: VGPR r, half hi
        S[row * NN + bj0 + c]      = fmaxf(acc0[r] + b0, 0.0f);
        S[row * NN + bj0 + 16 + c] = fmaxf(acc1[r] + b0, 0.0f);
    }
}

// In-place row softmax: one block per row, 4 elements per thread.
__global__ __launch_bounds__(256)
void GraphLearn_softmax_kernel(float* __restrict__ S)
{
    __shared__ float red[256];
    const int row = blockIdx.x;
    const int tid = threadIdx.x;
    float* rp = S + (size_t)row * NN;

    float v0 = rp[tid];
    float v1 = rp[tid + 256];
    float v2 = rp[tid + 512];
    float v3 = rp[tid + 768];

    float m = fmaxf(fmaxf(v0, v1), fmaxf(v2, v3));
    red[tid] = m;
    __syncthreads();
    for (int s = 128; s > 0; s >>= 1) {
        if (tid < s) red[tid] = fmaxf(red[tid], red[tid + s]);
        __syncthreads();
    }
    m = red[0];
    __syncthreads();

    float e0 = __expf(v0 - m);
    float e1 = __expf(v1 - m);
    float e2 = __expf(v2 - m);
    float e3 = __expf(v3 - m);

    red[tid] = (e0 + e1) + (e2 + e3);
    __syncthreads();
    for (int s = 128; s > 0; s >>= 1) {
        if (tid < s) red[tid] += red[tid + s];
        __syncthreads();
    }
    const float inv = 1.0f / red[0];

    rp[tid]       = e0 * inv;
    rp[tid + 256] = e1 * inv;
    rp[tid + 512] = e2 * inv;
    rp[tid + 768] = e3 * inv;
}

extern "C" void kernel_launch(void* const* d_in, const int* in_sizes, int n_in,
                              void* d_out, int out_size, void* d_ws, size_t ws_size,
                              hipStream_t stream)
{
    (void)in_sizes; (void)n_in; (void)d_ws; (void)ws_size; (void)out_size;
    const float* x    = (const float*)d_in[0];
    const float* wvec = (const float*)d_in[1];
    const float* wb   = (const float*)d_in[2];
    float* out = (float*)d_out;

    dim3 g1(NN / 32, NN / 64);   // (32, 16) = 512 blocks
    GraphLearn_scores_kernel<<<g1, dim3(128), 0, stream>>>(x, wvec, wb, out);
    GraphLearn_softmax_kernel<<<dim3(NN), dim3(256), 0, stream>>>(out);
}
